// BatchedMessagePassing_78091095376378
// MI455X (gfx1250) — compile-verified, matrix operand order checked by reading, not run on hardware
//
#include <hip/hip_runtime.h>
#include <hip/hip_bf16.h>

// ---- CDNA5 WMMA / TDM vector types ------------------------------------------
typedef __attribute__((ext_vector_type(16))) __bf16 v16bf;
typedef __attribute__((ext_vector_type(8)))  __bf16 v8bf;
typedef __attribute__((ext_vector_type(8)))  float  v8f;
typedef __attribute__((ext_vector_type(4)))  unsigned int v4u;
typedef __attribute__((ext_vector_type(8)))  int v8i;
typedef __attribute__((ext_vector_type(4)))  int v4i;

#define WSTRIDE 136   // padded bf16 row stride for W in LDS: 272B (16B aligned, bank-conflict free)
#define LN_EPS 1e-5f

// ---------------------------------------------------------------------------
// Kernel 0: zero the aggregation workspace (float4 grid-stride)
// ---------------------------------------------------------------------------
__global__ void zero_ws_kernel(float4* __restrict__ p, int n4) {
    int i      = blockIdx.x * blockDim.x + threadIdx.x;
    int stride = gridDim.x * blockDim.x;
    float4 z; z.x = 0.f; z.y = 0.f; z.z = 0.f; z.w = 0.f;
    for (; i < n4; i += stride) p[i] = z;
}

// ---------------------------------------------------------------------------
// Kernel 1: COO scatter  agg[l,r,:] += v * x[l,c,:]   (one wave32 per edge)
// ---------------------------------------------------------------------------
__global__ void spmm_scatter_kernel(const float* __restrict__ x,
                                    const int*   __restrict__ rows,
                                    const int*   __restrict__ cols,
                                    const float* __restrict__ vals,
                                    float* __restrict__ agg,
                                    int E, int G, int L) {
    int wave = (blockIdx.x * blockDim.x + threadIdx.x) >> 5;
    int lane = threadIdx.x & 31;
    if (wave >= E) return;

    int   r = rows[wave];
    int   c = cols[wave];
    float v = vals[wave];

    for (int l = 0; l < L; ++l) {
        const float4* src = (const float4*)(x + ((size_t)(l * G + c)) * 128);
        float*        dst = agg + ((size_t)(l * G + r)) * 128;
        float4 s = src[lane];                 // 32 lanes * 16B = full 128-float row
        int d = lane * 4;
        atomicAdd(dst + d + 0, v * s.x);      // global_atomic_add_f32, L2-resident
        atomicAdd(dst + d + 1, v * s.y);
        atomicAdd(dst + d + 2, v * s.z);
        atomicAdd(dst + d + 3, v * s.w);
    }
}

// ---------------------------------------------------------------------------
// Issue a TDM 1D copy: nElem fp32 from `gsrc` into LDS byte offset `lds_off`.
// D# per CDNA5 ISA (group0: count/lds/global/type,
// group1: data_size=4B, tensor_dim0 = tile_dim0 = nElem, stride0 = nElem).
// ---------------------------------------------------------------------------
__device__ __forceinline__ void tdm_load_1d(unsigned int lds_off,
                                            const void* gsrc,
                                            unsigned int nElem) {
    unsigned long long ga = (unsigned long long)gsrc;
    v4u g0;
    g0[0] = 1u;                                       // count=1, is_restore=0
    g0[1] = lds_off;                                  // lds_addr
    g0[2] = (unsigned int)(ga & 0xffffffffu);         // global_addr[31:0]
    g0[3] = (unsigned int)((ga >> 32) & 0x01ffffffu)  // global_addr[56:32]
          | (2u << 30);                               // type=2 ("image")
    v8i g1;
    g1[0] = (int)(2u << 16);                          // data_size=2 -> 4 bytes
    g1[1] = (int)((nElem & 0xffffu) << 16);           // tensor_dim0[15:0]
    g1[2] = (int)(((nElem >> 16) & 0xffffu))          // tensor_dim0[31:16]
          | (int)(1u << 16);                          // tensor_dim1 = 1
    g1[3] = (int)((nElem & 0xffffu) << 16);           // tile_dim0 = nElem
    g1[4] = 0;                                        // tile_dim1/2 unused
    g1[5] = (int)nElem;                               // tensor_dim0_stride lo
    g1[6] = 0;
    g1[7] = 0;
    v4i g2 = {0, 0, 0, 0};
    v4i g3 = {0, 0, 0, 0};
#if defined(__clang_major__) && (__clang_major__ >= 23)
    v8i gpad = {0, 0, 0, 0, 0, 0, 0, 0};
    __builtin_amdgcn_tensor_load_to_lds(g0, g1, g2, g3, gpad, 0);
#else
    __builtin_amdgcn_tensor_load_to_lds(g0, g1, g2, g3, 0);
#endif
}

// ---------------------------------------------------------------------------
// Kernel 2: GEMM (y @ W^T + b) + SiLU via v_wmma_f32_16x16x32_bf16.
// W staged global->LDS with the Tensor Data Mover in 4 x 16KB chunks, then
// converted to padded bf16. A (activations) split hi/lo bf16 (2 WMMAs/K-step).
// Each wave: 16 rows x 128 outputs. Block: 8 waves = 128 rows.
// ---------------------------------------------------------------------------
__launch_bounds__(256)
__global__ void gemm_silu_kernel(const float* __restrict__ agg,
                                 const float* __restrict__ W,
                                 const float* __restrict__ bias,
                                 float* __restrict__ out, int LG) {
    __shared__ struct {
        float  stage[32 * 128];        // 16 KB TDM staging (32 W rows)
        __bf16 w[128 * WSTRIDE];       // 34.8 KB padded bf16 W
    } lds;

    int tid = threadIdx.x;

    // ---- Stage W via TDM, chunk by chunk, converting fp32 -> bf16 in LDS ----
    unsigned int stage_off = (unsigned int)(unsigned long long)(&lds.stage[0]);
    for (int c = 0; c < 4; ++c) {
        if (tid < 32) {                               // wave 0 issues + waits
            tdm_load_1d(stage_off, W + c * 32 * 128, 32 * 128);
            __builtin_amdgcn_s_wait_tensorcnt(0);
        }
        __syncthreads();                              // chunk visible to block
        for (int i = tid; i < 32 * 128; i += 256) {
            int o = (c * 32) + (i >> 7), d = i & 127;
            lds.w[o * WSTRIDE + d] = (__bf16)lds.stage[i];
        }
        __syncthreads();                              // done before next chunk
    }

    int waveId = tid >> 5;
    int lane   = tid & 31;
    int g      = lane >> 4;     // K-half selector per ISA A/B fragment layout
    int n16    = lane & 15;     // M for A-frag loads, N for B/D columns
    int rowBase = blockIdx.x * 128 + waveId * 16;

    // ---- Load A fragments for 16 rows x 128 K, hi/lo bf16 split --------------
    v16bf ahi[4], alo[4];
    int arow = rowBase + n16;
    if (arow >= LG) arow = LG - 1;              // clamp (stores guarded below)
    const float* ap = agg + (size_t)arow * 128;
#pragma unroll
    for (int kt = 0; kt < 4; ++kt) {
        int k0 = kt * 32 + g * 8;
#pragma unroll
        for (int half = 0; half < 2; ++half) {
            const float4* p  = (const float4*)(ap + k0 + half * 16);
            float4 f0 = p[0], f1 = p[1];
            float fv[8] = {f0.x, f0.y, f0.z, f0.w, f1.x, f1.y, f1.z, f1.w};
#pragma unroll
            for (int e = 0; e < 8; ++e) {
                __bf16 h  = (__bf16)fv[e];
                float  rm = fv[e] - (float)h;
                ahi[kt][half * 8 + e] = h;
                alo[kt][half * 8 + e] = (__bf16)rm;
            }
        }
    }

    // ---- 8 output tiles of 16 columns each ----------------------------------
#pragma unroll
    for (int nt = 0; nt < 8; ++nt) {
        int   col  = nt * 16 + n16;
        float bcol = bias[col];
        v8f acc = {};
#pragma unroll
        for (int kt = 0; kt < 4; ++kt) {
            // B fragment: B[k][n] = W[n][k]; 2x 16B LDS loads, conflict-free.
            const __bf16* wp = &lds.w[col * WSTRIDE + kt * 32 + g * 8];
            v8bf b0 = *(const v8bf*)(wp);
            v8bf b1 = *(const v8bf*)(wp + 16);
            v16bf bh = __builtin_shufflevector(b0, b1,
                0, 1, 2, 3, 4, 5, 6, 7, 8, 9, 10, 11, 12, 13, 14, 15);
            acc = __builtin_amdgcn_wmma_f32_16x16x32_bf16(
                      false, ahi[kt], false, bh, (short)0, acc, false, false);
            acc = __builtin_amdgcn_wmma_f32_16x16x32_bf16(
                      false, alo[kt], false, bh, (short)0, acc, false, false);
        }
        // Epilogue: +bias, SiLU, store. D-frag layout: VGPR r -> row r + 8*g.
#pragma unroll
        for (int r = 0; r < 8; ++r) {
            int row = rowBase + r + g * 8;
            if (row < LG) {
                float h = acc[r] + bcol;
                float s = h * (1.0f / (1.0f + __expf(-h)));
                out[(size_t)row * 128 + col] = s;
            }
        }
    }
}

// ---------------------------------------------------------------------------
// Kernel 3: LayerNorm in-place on d_out; one wave32 per 128-float row.
// ---------------------------------------------------------------------------
__global__ void layernorm_kernel(float* __restrict__ out,
                                 const float* __restrict__ gamma,
                                 const float* __restrict__ beta, int LG) {
    int wave = (blockIdx.x * blockDim.x + threadIdx.x) >> 5;
    int lane = threadIdx.x & 31;
    if (wave >= LG) return;

    float* p = out + (size_t)wave * 128;
    float v[4];
    float sum = 0.f, sq = 0.f;
#pragma unroll
    for (int j = 0; j < 4; ++j) {
        v[j] = p[lane + 32 * j];
        sum += v[j];
        sq  += v[j] * v[j];
    }
#pragma unroll
    for (int m = 16; m >= 1; m >>= 1) {
        sum += __shfl_xor(sum, m, 32);
        sq  += __shfl_xor(sq,  m, 32);
    }
    float mu  = sum * (1.0f / 128.0f);
    float var = sq  * (1.0f / 128.0f) - mu * mu;
    float inv = rsqrtf(var + LN_EPS);
#pragma unroll
    for (int j = 0; j < 4; ++j) {
        int d = lane + 32 * j;
        p[d] = (v[j] - mu) * inv * gamma[d] + beta[d];
    }
}

// ---------------------------------------------------------------------------
extern "C" void kernel_launch(void* const* d_in, const int* in_sizes, int n_in,
                              void* d_out, int out_size, void* d_ws, size_t ws_size,
                              hipStream_t stream) {
    const float* x     = (const float*)d_in[0];
    const int*   rows  = (const int*)  d_in[1];
    const int*   cols  = (const int*)  d_in[2];
    const float* vals  = (const float*)d_in[3];
    const float* W     = (const float*)d_in[4];
    const float* b     = (const float*)d_in[5];
    const float* gamma = (const float*)d_in[6];
    const float* beta  = (const float*)d_in[7];

    const int D  = in_sizes[5];        // 128
    const int E  = in_sizes[1];        // 800000
    const int LG = in_sizes[0] / D;    // L*G = 100000
    const int L  = 2;                  // reference fixes L = 2
    const int G  = LG / L;

    float* agg = (float*)d_ws;         // LG * D floats (51.2 MB)
    int n4 = (LG * D) / 4;
    int zb = (n4 + 255) / 256; if (zb > 4096) zb = 4096;

    zero_ws_kernel     <<<zb,               256, 0, stream>>>((float4*)agg, n4);
    spmm_scatter_kernel<<<(E + 7) / 8,      256, 0, stream>>>(x, rows, cols, vals, agg, E, G, L);
    gemm_silu_kernel   <<<(LG + 127) / 128, 256, 0, stream>>>(agg, W, b, (float*)d_out, LG);
    layernorm_kernel   <<<(LG + 7) / 8,     256, 0, stream>>>((float*)d_out, gamma, beta, LG);
}
